// AttentionNet_19696720019836
// MI455X (gfx1250) — matrix operand order repaired and checked by reading, not compiled, not run on hardware
//
#include <hip/hip_runtime.h>

// ---------- types ----------
typedef __attribute__((ext_vector_type(16))) __bf16 v16bf;
typedef __attribute__((ext_vector_type(8)))  float  v8f;
typedef __attribute__((ext_vector_type(8)))  unsigned short v8us;
typedef __attribute__((ext_vector_type(16))) unsigned short v16us;
typedef long long ll;

#define MODE_BF16      0
#define MODE_BF16_T    1
#define MODE_F32       2
#define MODE_BF16_SILU 3

__device__ __forceinline__ unsigned short f2bf(float f) {
  unsigned int u = __float_as_uint(f);
  u += 0x7FFFu + ((u >> 16) & 1u);          // round-to-nearest-even
  return (unsigned short)(u >> 16);
}

__device__ __forceinline__ v16bf frag_from(const unsigned short* p0,
                                           const unsigned short* p1) {
  v8us a = *(const v8us*)p0;                // ds_load_b128
  v8us b = *(const v8us*)p1;                // ds_load_b128
  v16us w = __builtin_shufflevector(a, b, 0,1,2,3,4,5,6,7,8,9,10,11,12,13,14,15);
  return __builtin_bit_cast(v16bf, w);
}

// ---- CDNA5 async global->LDS copy (ASYNCcnt path), portable via inline asm ----
__device__ __forceinline__ void async_ld16(unsigned lds_addr, const void* gptr) {
  asm volatile("global_load_async_to_lds_b128 %0, %1, off"
               :: "v"(lds_addr), "v"((unsigned long long)(uintptr_t)gptr)
               : "memory");
}
__device__ __forceinline__ void wait_async_le6() {
  asm volatile("s_wait_asynccnt 0x6" ::: "memory");
}
__device__ __forceinline__ void wait_async_0() {
  asm volatile("s_wait_asynccnt 0x0" ::: "memory");
}

// ---------- bf16 TN GEMM: C[M,N] = A[M,K] * B[N,K]^T  (lda=ldb=K) ----------
#define BM 128
#define BN 256
#define BK 32
#define PITCH 40   // halves; 80B row pitch -> 16B aligned, bank-friendly

__global__ __launch_bounds__(256) void gemm_bf16_tn(
    const unsigned short* __restrict__ A, const unsigned short* __restrict__ B,
    void* __restrict__ Cv, int M, int N, int K,
    ll sA, ll sB, ll sC, float scale, int mode)
{
  // double-buffered tiles: 2*(128+256)*40*2B = 60KB LDS
  __shared__ __align__(16) unsigned short lA[2][BM * PITCH];
  __shared__ __align__(16) unsigned short lB[2][BN * PITCH];

  A += (ll)blockIdx.z * sA;
  B += (ll)blockIdx.z * sB;

  const int tid  = threadIdx.x;
  const int lane = tid & 31;
  const int wave = tid >> 5;          // 8 waves
  const int lg   = lane >> 4;         // lane group 0/1
  const int lr   = lane & 15;
  const int wm   = wave >> 2;         // 0..1 -> 64 rows each
  const int wn   = wave & 3;          // 0..3 -> 64 cols each
  const int m0   = blockIdx.y * BM;
  const int n0   = blockIdx.x * BN;
  // tile-load mapping: A rows by thread pairs, B one row per thread
  const int grA  = tid >> 1;          // 0..127
  const int gcA  = (tid & 1) << 4;    // 0 or 16 halves

  // LDS byte addresses for async destinations (addr[31:0] = LDS offset)
  unsigned stA[2], stB[2];
  stA[0] = (unsigned)(uintptr_t)&lA[0][grA * PITCH + gcA];
  stA[1] = (unsigned)(uintptr_t)&lA[1][grA * PITCH + gcA];
  stB[0] = (unsigned)(uintptr_t)&lB[0][tid * PITCH];
  stB[1] = (unsigned)(uintptr_t)&lB[1][tid * PITCH];

  v8f acc[4][4] = {};

  const unsigned short* ga = A + (ll)(m0 + grA) * K + gcA;
  const unsigned short* gb = B + (ll)(n0 + tid) * K;
  const int Kt = K / BK;

  // prime buffer 0: A 2 chunks + B 4 chunks per thread (6 async copies)
  async_ld16(stA[0], ga);       async_ld16(stA[0] + 16, ga + 8);
  async_ld16(stB[0], gb);       async_ld16(stB[0] + 16, gb + 8);
  async_ld16(stB[0] + 32, gb + 16); async_ld16(stB[0] + 48, gb + 24);
  ga += BK; gb += BK;

  int p = 0;
  for (int kt = 0; kt < Kt; ++kt) {
    if (kt + 1 < Kt) {
      // issue next tile into the other buffer, then wait for current tile only
      unsigned sa = stA[p ^ 1], sb = stB[p ^ 1];
      async_ld16(sa, ga);           async_ld16(sa + 16, ga + 8);
      async_ld16(sb, gb);           async_ld16(sb + 16, gb + 8);
      async_ld16(sb + 32, gb + 16); async_ld16(sb + 48, gb + 24);
      if (kt + 2 < Kt) {                     // warm L2 two tiles ahead
        __builtin_prefetch(ga + BK, 0, 3);
        __builtin_prefetch(gb + BK, 0, 3);
      }
      ga += BK; gb += BK;
      wait_async_le6();                      // current tile complete (in-order)
    } else {
      wait_async_0();
    }
    __syncthreads();                         // all waves' copies visible

    const unsigned short* lap = lA[p];
    const unsigned short* lbp = lB[p];
    v16bf af[4], bfr[4];
#pragma unroll
    for (int mi = 0; mi < 4; ++mi) {
      // A 16x32 frag: lane grp0: K0-7 & 16-23, grp1: K8-15 & 24-31
      const unsigned short* q = &lap[(wm * 64 + mi * 16 + lr) * PITCH + lg * 8];
      af[mi] = frag_from(q, q + 16);
    }
#pragma unroll
    for (int ni = 0; ni < 4; ++ni) {
      // B 32x16 frag: lane grp0: K0-7 & 16-23, grp1: K8-15 & 24-31 (column = lr)
      const unsigned short* q = &lbp[(wn * 64 + ni * 16 + lr) * PITCH + lg * 8];
      bfr[ni] = frag_from(q, q + 16);
    }
#pragma unroll
    for (int mi = 0; mi < 4; ++mi)
#pragma unroll
      for (int ni = 0; ni < 4; ++ni)
        acc[mi][ni] = __builtin_amdgcn_wmma_f32_16x16x32_bf16(
            false, af[mi], false, bfr[ni], (short)0, acc[mi][ni], false, false);

    __syncthreads();                         // done reading buf p
    p ^= 1;
  }

  unsigned short* Cb = (unsigned short*)Cv + (ll)blockIdx.z * sC;
  float*          Cf = (float*)Cv          + (ll)blockIdx.z * sC;

#pragma unroll
  for (int mi = 0; mi < 4; ++mi)
#pragma unroll
    for (int ni = 0; ni < 4; ++ni) {
      const int mb = m0 + wm * 64 + mi * 16 + lg * 8;  // VGPR r -> row mb+r
      const int nb = n0 + wn * 64 + ni * 16 + lr;
#pragma unroll
      for (int r = 0; r < 8; ++r) {
        const int m = mb + r;
        float v = acc[mi][ni][r];
        if (mode == MODE_F32) {
          Cf[(ll)m * N + nb] = v * scale;
        } else if (mode == MODE_BF16) {
          Cb[(ll)m * N + nb] = f2bf(v);
        } else if (mode == MODE_BF16_T) {
          Cb[(ll)nb * M + m] = f2bf(v);           // transposed store (for V^T)
        } else {                                  // silu epilogue
          float s = v / (1.f + __expf(-v));
          Cb[(ll)m * N + nb] = f2bf(s);
        }
      }
    }
}

// ---------- row softmax: f32 in; bf16 or f32 out ----------
__global__ __launch_bounds__(256) void softmax_rows(
    const float* __restrict__ in, void* __restrict__ outv, int len, int out_bf16)
{
  __shared__ float red[256];
  const int t = threadIdx.x;
  const float* x = in + (ll)blockIdx.x * len;

  float mx = -1e30f;
  for (int i = t; i < len; i += 256) mx = fmaxf(mx, x[i]);
  red[t] = mx; __syncthreads();
  for (int s = 128; s > 0; s >>= 1) {
    if (t < s) red[t] = fmaxf(red[t], red[t + s]);
    __syncthreads();
  }
  mx = red[0]; __syncthreads();

  float sum = 0.f;
  for (int i = t; i < len; i += 256) sum += __expf(x[i] - mx);
  red[t] = sum; __syncthreads();
  for (int s = 128; s > 0; s >>= 1) {
    if (t < s) red[t] += red[t + s];
    __syncthreads();
  }
  const float inv = 1.f / red[0];

  if (out_bf16) {
    unsigned short* o = (unsigned short*)outv + (ll)blockIdx.x * len;
    for (int i = t; i < len; i += 256) o[i] = f2bf(__expf(x[i] - mx) * inv);
  } else {
    float* o = (float*)outv + (ll)blockIdx.x * len;
    for (int i = t; i < len; i += 256) o[i] = __expf(x[i] - mx) * inv;
  }
}

// ---------- f32 -> bf16 convert ----------
__global__ __launch_bounds__(256) void f32_to_bf16(
    const float* __restrict__ in, unsigned short* __restrict__ out, int n)
{
  int i = blockIdx.x * 256 + threadIdx.x;
  if (i < n) out[i] = f2bf(in[i]);
}

// ---------- orchestration ----------
extern "C" void kernel_launch(void* const* d_in, const int* in_sizes, int n_in,
                              void* d_out, int out_size, void* d_ws, size_t ws_size,
                              hipStream_t stream) {
  (void)in_sizes; (void)n_in; (void)out_size; (void)ws_size;
  const int Bq = 4, S = 2048, D = 1024, Mall = Bq * S;
  const float inv_scale = 0.08838834764831845f;   // 1/sqrt(128)

  unsigned char* ws = (unsigned char*)d_ws;
  // workspace layout (bytes)
  const size_t oW  = 0;                              // 8 bf16 weights: 16MB
  const size_t oXb = oW  + (size_t)8 * D * D * 2;    // x bf16: 16MB
  const size_t oQ  = oXb + (size_t)Mall * D * 2;
  const size_t oK  = oQ  + (size_t)Mall * D * 2;
  const size_t oVt = oK  + (size_t)Mall * D * 2;
  const size_t oP  = oVt + (size_t)Mall * D * 2;     // probs bf16: 32MB
  const size_t oH  = oP  + (size_t)Bq * S * S * 2;   // silu(attn) bf16: 16MB
  const size_t oS  = oH  + (size_t)Mall * D * 2;     // scores/logits f32: 64MB

  auto Wb = [&](int j) { return (unsigned short*)(ws + oW + (size_t)j * D * D * 2); };
  unsigned short* xb     = (unsigned short*)(ws + oXb);
  unsigned short* Qb     = (unsigned short*)(ws + oQ);
  unsigned short* Kb     = (unsigned short*)(ws + oK);
  unsigned short* Vt     = (unsigned short*)(ws + oVt);
  unsigned short* probs  = (unsigned short*)(ws + oP);
  unsigned short* hb     = (unsigned short*)(ws + oH);
  float*          scores = (float*)(ws + oS);

  auto gemm = [&](const void* A, const void* Bm, void* C, int M, int N, int K,
                  ll sA, ll sB, ll sC, int batch, float sc, int mode) {
    dim3 g(N / BN, M / BM, batch);
    gemm_bf16_tn<<<g, 256, 0, stream>>>((const unsigned short*)A,
                                        (const unsigned short*)Bm, C,
                                        M, N, K, sA, sB, sC, sc, mode);
  };

  // convert inputs to bf16
  f32_to_bf16<<<(Mall * D) / 256, 256, 0, stream>>>((const float*)d_in[0], xb, Mall * D);
  for (int j = 0; j < 8; ++j)
    f32_to_bf16<<<(D * D) / 256, 256, 0, stream>>>((const float*)d_in[1 + j], Wb(j), D * D);

  auto run_layer = [&](const unsigned short* xin, int jq, int jk, int jv, int jfc,
                       bool final_layer) {
    // projections (batch-independent): [8192,1024] x [1024,1024]^T
    gemm(xin, Wb(jq), Qb, Mall, D, D, 0, 0, 0, 1, 1.f, MODE_BF16);
    gemm(xin, Wb(jk), Kb, Mall, D, D, 0, 0, 0, 1, 1.f, MODE_BF16);
    // V projection, stored transposed per batch: Vt[b] = [D,S]
    gemm(xin, Wb(jv), Vt, S, D, D, (ll)S * D, 0, (ll)D * S, Bq, 1.f, MODE_BF16_T);
    // scores[b] = Q[b] K[b]^T / sqrt(128)  -> f32
    gemm(Qb, Kb, scores, S, S, D, (ll)S * D, (ll)S * D, (ll)S * S, Bq,
         inv_scale, MODE_F32);
    // softmax rows -> bf16 probs
    softmax_rows<<<Mall, 256, 0, stream>>>(scores, probs, S, 1);
    // O[b] = probs[b] (2048x2048) * Vt[b]^T (TN, K=2048); fused silu -> bf16
    gemm(probs, Vt, hb, S, D, S, (ll)S * S, (ll)D * S, (ll)S * D, Bq,
         1.f, MODE_BF16_SILU);
    // fc: h @ W^T
    if (!final_layer)
      gemm(hb, Wb(jfc), xb, Mall, D, D, 0, 0, 0, 1, 1.f, MODE_BF16);  // reuse xb
    else
      gemm(hb, Wb(jfc), scores, Mall, D, D, 0, 0, 0, 1, 1.f, MODE_F32); // logits
  };

  run_layer(xb, 0, 1, 2, 3, false);
  run_layer(xb, 4, 5, 6, 7, true);

  // final softmax over D -> f32 output
  softmax_rows<<<Mall, 256, 0, stream>>>(scores, d_out, D, 0);
}
